// GatedQuestionAnswering_19456201851434
// MI455X (gfx1250) — matrix operand order repaired and checked by reading, not compiled, not run on hardware
//
#include <hip/hip_runtime.h>
#include <math.h>

// ---------------------------------------------------------------------------
// GatedQuestionAnswering for MI455X (gfx1250, wave32, WMMA).
//
// Output = dec_hs[1024,1024] @ W_pred[1024,28996]  (~60 GFLOP, ~180MB traffic
// with bf16 weights -> memory bound at 23.3 TB/s). Strategy:
//   * all GEMMs in bf16 with f32 accumulation via v_wmma_f32_16x16x32_bf16
//     (wave tile 16x64: one A fragment feeds 4 WMMAs, branchless K-loop)
//   * batched input projections (wih @ x for every timestep) as WMMA GEMMs
//   * sequential GRU recurrences as persistent-grid kernels with the whh
//     weight slice resident in LDS (bf16); per-step hidden-state broadcast
//     refreshed via GLOBAL_LOAD_ASYNC_TO_LDS_B128 (+ s_wait_asynccnt)
//   * backward encoder reduces algebraically to ONE cell (h0 == 0 and only
//     bwd_hs[-1] is consumed), forward encoder only needs its final state.
// ---------------------------------------------------------------------------

typedef __attribute__((ext_vector_type(16))) __bf16 v16bf;
typedef __attribute__((ext_vector_type(8)))  float  v8f;

#define HID   512
#define BIH   1024
#define INSZ  768
#define LIN   512
#define LOUT  1024
#define VOCAB 28996

// ---- bf16 <-> f32 as raw bits (no scalar __bf16 arithmetic needed) --------
__device__ __forceinline__ unsigned short f2bf(float f) {
  unsigned u = __float_as_uint(f);
  unsigned r = u + 0x7FFFu + ((u >> 16) & 1u);   // round-to-nearest-even
  return (unsigned short)(r >> 16);
}
__device__ __forceinline__ float bf2f(unsigned short s) {
  return __uint_as_float(((unsigned)s) << 16);
}

// ---------------------------------------------------------------------------
__global__ void k_init(float* h, unsigned* bar) {
  int i = blockIdx.x * blockDim.x + threadIdx.x;
  if (i < HID) h[i] = 0.f;     // fw encoder h0
  if (i < 8)   bar[i] = 0u;    // grid-barrier state (counter, generation)
}

__global__ void k_f32_to_bf16(const float* __restrict__ s,
                              unsigned short* __restrict__ d, size_t n) {
  size_t i = (size_t)blockIdx.x * blockDim.x + threadIdx.x;
  size_t stride = (size_t)gridDim.x * blockDim.x;
  for (; i < n; i += stride) d[i] = f2bf(s[i]);
}

// src [K,N] f32 row-major -> dst [N,K] bf16 row-major (LDS-tiled transpose)
__global__ void k_transpose_to_bf16(const float* __restrict__ src,
                                    unsigned short* __restrict__ dst,
                                    int K, int N) {
  __shared__ float tile[32][33];
  int kb = blockIdx.y * 32, nb = blockIdx.x * 32;
  int tx = threadIdx.x, ty = threadIdx.y;        // block 32x8
  for (int yy = ty; yy < 32; yy += 8) {
    int k = kb + yy, n = nb + tx;
    tile[yy][tx] = (k < K && n < N) ? src[(size_t)k * N + n] : 0.f;
  }
  __syncthreads();
  for (int yy = ty; yy < 32; yy += 8) {
    int n = nb + yy, k = kb + tx;
    if (n < N && k < K) dst[(size_t)n * K + k] = f2bf(tile[tx][yy]);
  }
}

// ---------------------------------------------------------------------------
// C[M,N] f32 = A[M,K](bf16, row-major) * B[N,K](bf16, row-major, i.e. B^T).
// 256 threads = 8 waves in a 4(M) x 2(N) grid; wave tile 16(M) x 64(N);
// block macro-tile 64 x 128. Branchless K-loop: OOB columns clamp their load
// row (their accumulators are garbage but never stored).
// A/B fragments use contiguous K (k = 16*(lane/16) + s); C/D per CDNA5 ISA 7.12.2 (wave32).
// ---------------------------------------------------------------------------
__global__ void __launch_bounds__(256)
k_gemm_bf16_bt(const unsigned short* __restrict__ A,
               const unsigned short* __restrict__ B,
               float* __restrict__ C, int M, int N, int K) {
  const int lane = threadIdx.x & 31;
  const int wave = threadIdx.x >> 5;
  const int wm = wave >> 1, wn = wave & 1;             // 4 x 2 wave grid
  const int m0 = blockIdx.y * 64 + wm * 16;
  const int n0 = blockIdx.x * 128 + wn * 64;

  const int row   = m0 + (lane & 15);                  // A row for this lane
  const int kbA   = (lane < 16) ? 0 : 16;              // A frag K-phase
  const int koffB = (lane < 16) ? 0 : 16;              // B frag K-phase
  const int ncol  = lane & 15;

  // Clamp OOB columns to a valid row so loads never branch.
  int nb[4];
  for (int j = 0; j < 4; ++j) {
    int c = n0 + ncol + 16 * j;
    nb[j] = (c < N) ? c : (N - 1);
  }

  v8f acc[4] = {{}, {}, {}, {}};
  union Frag { v16bf v; uint4 q[2]; };

  const size_t arow = (size_t)row * K;
  size_t brow[4];
  for (int j = 0; j < 4; ++j) brow[j] = (size_t)nb[j] * K;

  for (int k0 = 0; k0 < K; k0 += 32) {
    Frag a;
    // A: elements 0..15 = K[k0+kb .. +15] (contiguous, matches B)
    a.q[0] = *reinterpret_cast<const uint4*>(A + arow + k0 + kbA);
    a.q[1] = *reinterpret_cast<const uint4*>(A + arow + k0 + kbA + 8);

    Frag b[4];
#pragma unroll
    for (int j = 0; j < 4; ++j) {
      // B: 16 contiguous bf16 along K for this lane's column
      const uint4* pb = reinterpret_cast<const uint4*>(B + brow[j] + k0 + koffB);
      b[j].q[0] = pb[0];
      b[j].q[1] = pb[1];
    }

    if (k0 + 32 < K) {                                  // global_prefetch_b8
      __builtin_prefetch(A + arow + k0 + 32 + kbA, 0, 3);
      __builtin_prefetch(B + brow[0] + k0 + 32 + koffB, 0, 3);
      __builtin_prefetch(B + brow[2] + k0 + 32 + koffB, 0, 3);
    }

#pragma unroll
    for (int j = 0; j < 4; ++j)
      acc[j] = __builtin_amdgcn_wmma_f32_16x16x32_bf16(
          false, a.v, false, b[j].v, (short)0, acc[j], false, false);
  }

  // C/D layout: lanes 0-15 -> N=lane, M=m0+r ; lanes 16-31 -> N=lane-16, M=m0+8+r
  const int mst = m0 + ((lane >> 4) << 3);
#pragma unroll
  for (int j = 0; j < 4; ++j) {
    int col = n0 + ncol + 16 * j;
    if (col < N) {
      for (int r = 0; r < 8; ++r)
        C[(size_t)(mst + r) * N + col] = acc[j][r];
    }
  }
}

// ---------------------------------------------------------------------------
// Backward encoder collapses to a single GRU cell on input row L-1 with h0=0.
// ---------------------------------------------------------------------------
__global__ void k_bw_single_cell(const float* __restrict__ x,     // [INSZ]
                                 const float* __restrict__ wih,   // [3H, INSZ]
                                 const float* __restrict__ bih,
                                 const float* __restrict__ bhh,
                                 float* __restrict__ hout) {      // [HID]
  int i = blockIdx.x * blockDim.x + threadIdx.x;
  if (i >= HID) return;
  float gr = 0.f, gz = 0.f, gn = 0.f;
  const float* wr = wih + (size_t)i * INSZ;
  const float* wz = wih + (size_t)(HID + i) * INSZ;
  const float* wn = wih + (size_t)(2 * HID + i) * INSZ;
  for (int k = 0; k < INSZ; ++k) {
    float xv = x[k];
    gr = fmaf(wr[k], xv, gr);
    gz = fmaf(wz[k], xv, gz);
    gn = fmaf(wn[k], xv, gn);
  }
  float r = 1.f / (1.f + expf(-(gr + bih[i] + bhh[i])));
  float z = 1.f / (1.f + expf(-(gz + bih[HID + i] + bhh[HID + i])));
  float n = tanhf(gn + bih[2 * HID + i] + r * bhh[2 * HID + i]);
  hout[i] = (1.f - z) * n;                          // + z * 0
}

// ---------------------------------------------------------------------------
// Grid barrier (persistent-kernel synchronization via L2 atomics + s_sleep)
// ---------------------------------------------------------------------------
__device__ __forceinline__ void grid_barrier(unsigned* counter,
                                             volatile unsigned* gen,
                                             unsigned nblk) {
  __syncthreads();
  if (threadIdx.x == 0) {
    __threadfence();
    unsigned g = *gen;
    if (atomicAdd(counter, 1u) == nblk - 1u) {
      *counter = 0u;
      __threadfence();
      atomicAdd((unsigned*)gen, 1u);
    } else {
      while (*gen == g) __builtin_amdgcn_s_sleep(2);
    }
    __threadfence();
  }
  __syncthreads();
}

// ---------------------------------------------------------------------------
// Persistent GRU recurrence. gridDim.x * 24 == 3H, gridDim.x * 8 == H.
// Each block keeps a 24-row slice of whh in LDS as bf16 for the whole
// sequence. The per-step hidden-state broadcast (global h -> LDS) uses the
// CDNA5 async DMA path: GLOBAL_LOAD_ASYNC_TO_LDS_B128 + s_wait_asynccnt.
//   fw encoder : 64 blocks,  H=512,  L=512,  shift_input=0, hs_bf=null
//   decoder    : 128 blocks, H=1024, L=1024, shift_input=1, hs_bf=dec_hs
// ---------------------------------------------------------------------------
__global__ void __launch_bounds__(256)
k_gru_persistent(const float* __restrict__ gx,    // [Lg, 3H] = wih @ x (no bias)
                 const float* __restrict__ whh,   // [3H, H] f32
                 const float* __restrict__ bih,
                 const float* __restrict__ bhh,
                 float* __restrict__ h,           // [H] state (pre-initialized)
                 float* __restrict__ ghbuf,       // [3H] scratch
                 unsigned short* hs_bf,           // [L, H] bf16 out, or null
                 int H, int L, int shift_input, unsigned* bar) {
  extern __shared__ char smem[];
  unsigned short* w_s = (unsigned short*)smem;                  // [24*H] bf16
  float* h_s = (float*)(smem + (size_t)24 * H * sizeof(unsigned short));
  const unsigned NB = gridDim.x;
  const int tid = threadIdx.x, lane = tid & 31, wave = tid >> 5;
  const int rowBase = blockIdx.x * 24;
  const int H3 = 3 * H;

  // Stage this block's whh slice into LDS (f32 -> bf16 once).
  for (int idx = tid; idx < 24 * H; idx += 256) {
    int rl = idx / H, k = idx - rl * H;
    w_s[idx] = f2bf(whh[(size_t)(rowBase + rl) * H + k]);
  }
  __syncthreads();

  // Async DMA setup for the per-step h broadcast: dynamic LDS starts at
  // offset 0 (no static __shared__ in this kernel), so the LDS byte address
  // of h_s is simply 24*H*2.  Each of the first H/4 threads moves 16 bytes.
  const unsigned h_lds_addr = (unsigned)(24 * H * 2) + (unsigned)tid * 16u;
  const float* h_gsrc = h + tid * 4;
  const bool do_dma = tid < (H >> 2);

  const int kPerLane = H >> 5;           // 16 (enc) or 32 (dec)
  const int k0 = lane * kPerLane;        // contiguous K chunk per lane

  for (int t = 0; t < L; ++t) {
    if (do_dma) {
      asm volatile("global_load_async_to_lds_b128 %0, %1, off"
                   :: "v"(h_lds_addr), "v"(h_gsrc) : "memory");
    }
    asm volatile("s_wait_asynccnt 0x0" ::: "memory");
    __syncthreads();
    const float* gxrow =
        gx + (size_t)(shift_input ? (t ? t - 1 : 0) : t) * H3;

    // gh[j] = whh[j,:] . h  for this block's 24 rows (3 rows per wave)
    for (int rr = 0; rr < 3; ++rr) {
      int rl = wave * 3 + rr;
      const unsigned short* wr = w_s + (size_t)rl * H + k0;
      const float* hp = h_s + k0;
      float acc = 0.f;
#pragma unroll 8
      for (int kk = 0; kk < kPerLane; ++kk)
        acc = fmaf(bf2f(wr[kk]), hp[kk], acc);
      for (int off = 16; off > 0; off >>= 1)
        acc += __shfl_down(acc, off, 32);
      if (lane == 0) ghbuf[rowBase + rl] = acc;
    }
    grid_barrier(bar, bar + 1, NB);

    // gate math for this block's 8 hidden units
    if (tid < 8) {
      int i = blockIdx.x * 8 + tid;
      float hr = ghbuf[i] + bhh[i];
      float hz = ghbuf[H + i] + bhh[H + i];
      float hn = ghbuf[2 * H + i] + bhh[2 * H + i];
      float r = 1.f / (1.f + expf(-(gxrow[i] + bih[i] + hr)));
      float z = 1.f / (1.f + expf(-(gxrow[H + i] + bih[H + i] + hz)));
      float n = tanhf(gxrow[2 * H + i] + bih[2 * H + i] + r * hn);
      float hnew = (1.f - z) * n + z * h[i];
      h[i] = hnew;
      if (hs_bf) hs_bf[(size_t)t * H + i] = f2bf(hnew);
    }
    grid_barrier(bar, bar + 1, NB);
  }
}

// ---------------------------------------------------------------------------
extern "C" void kernel_launch(void* const* d_in, const int* in_sizes, int n_in,
                              void* d_out, int out_size, void* d_ws,
                              size_t ws_size, hipStream_t stream) {
  const float* input_ctx  = (const float*)d_in[0];
  const float* output_ctx = (const float*)d_in[1];
  const float* fw_wih = (const float*)d_in[2];
  const float* fw_whh = (const float*)d_in[3];
  const float* fw_bih = (const float*)d_in[4];
  const float* fw_bhh = (const float*)d_in[5];
  const float* bw_wih = (const float*)d_in[6];
  // d_in[7] = bw_whh: unused (h0 == 0 -> whh @ h0 == 0)
  const float* bw_bih = (const float*)d_in[8];
  const float* bw_bhh = (const float*)d_in[9];
  const float* dec_wih = (const float*)d_in[10];
  const float* dec_whh = (const float*)d_in[11];
  const float* dec_bih = (const float*)d_in[12];
  const float* dec_bhh = (const float*)d_in[13];
  const float* W_pred  = (const float*)d_in[14];
  float* preds = (float*)d_out;

  char* ws = (char*)d_ws;
  size_t off = 0;
  auto alloc = [&](size_t bytes) -> void* {
    void* p = ws + off;
    off = (off + bytes + 255) & ~(size_t)255;
    return p;
  };
  unsigned short* in_bf     = (unsigned short*)alloc((size_t)LIN * INSZ * 2);
  unsigned short* outc_bf   = (unsigned short*)alloc((size_t)LOUT * INSZ * 2);
  unsigned short* fwih_bf   = (unsigned short*)alloc((size_t)3 * HID * INSZ * 2);
  unsigned short* dwih_bf   = (unsigned short*)alloc((size_t)3 * BIH * INSZ * 2);
  unsigned short* wpredT_bf = (unsigned short*)alloc((size_t)VOCAB * BIH * 2);
  float* gx_fw  = (float*)alloc((size_t)LIN * 3 * HID * 4);
  float* gx_dec = (float*)alloc((size_t)LOUT * 3 * BIH * 4);
  float* hbuf   = (float*)alloc(BIH * 4);
  float* ghb    = (float*)alloc(3 * BIH * 4);
  unsigned short* dec_hs_bf = (unsigned short*)alloc((size_t)LOUT * BIH * 2);
  unsigned* bar = (unsigned*)alloc(256);

  // 0) init state + barrier
  k_init<<<2, 256, 0, stream>>>(hbuf, bar);

  // 1) f32 -> bf16 conversions (weights already [N,K] row-major)
  size_t n;
  n = (size_t)LIN * INSZ;
  k_f32_to_bf16<<<(unsigned)((n + 255) / 256), 256, 0, stream>>>(input_ctx, in_bf, n);
  n = (size_t)LOUT * INSZ;
  k_f32_to_bf16<<<(unsigned)((n + 255) / 256), 256, 0, stream>>>(output_ctx, outc_bf, n);
  n = (size_t)3 * HID * INSZ;
  k_f32_to_bf16<<<(unsigned)((n + 255) / 256), 256, 0, stream>>>(fw_wih, fwih_bf, n);
  n = (size_t)3 * BIH * INSZ;
  k_f32_to_bf16<<<(unsigned)((n + 255) / 256), 256, 0, stream>>>(dec_wih, dwih_bf, n);
  // W_pred [BIH, VOCAB] -> [VOCAB, BIH] bf16
  k_transpose_to_bf16<<<dim3((VOCAB + 31) / 32, BIH / 32), dim3(32, 8), 0, stream>>>(
      W_pred, wpredT_bf, BIH, VOCAB);

  // 2) batched input projections (WMMA)
  k_gemm_bf16_bt<<<dim3(3 * HID / 128, LIN / 64), 256, 0, stream>>>(
      in_bf, fwih_bf, gx_fw, LIN, 3 * HID, INSZ);
  k_gemm_bf16_bt<<<dim3(3 * BIH / 128, LOUT / 64), 256, 0, stream>>>(
      outc_bf, dwih_bf, gx_dec, LOUT, 3 * BIH, INSZ);

  // 3) backward encoder == one cell -> h[HID..BIH)
  k_bw_single_cell<<<2, 256, 0, stream>>>(
      input_ctx + (size_t)(LIN - 1) * INSZ, bw_wih, bw_bih, bw_bhh, hbuf + HID);

  // 4) forward encoder recurrence -> h[0..HID)
  k_gru_persistent<<<64, 256, 24 * HID * 2 + HID * 4, stream>>>(
      gx_fw, fw_whh, fw_bih, fw_bhh, hbuf, ghb, nullptr, HID, LIN, 0, bar);

  // 5) decoder recurrence -> dec_hs (bf16)
  k_gru_persistent<<<128, 256, 24 * BIH * 2 + BIH * 4, stream>>>(
      gx_dec, dec_whh, dec_bih, dec_bhh, hbuf, ghb, dec_hs_bf, BIH, LOUT, 1, bar);

  // 6) vocab projection (WMMA): preds = dec_hs @ W_pred
  k_gemm_bf16_bt<<<dim3((VOCAB + 127) / 128, LOUT / 64), 256, 0, stream>>>(
      dec_hs_bf, wpredT_bf, preds, LOUT, VOCAB, BIH);
}
